// AlbertDecoderAttention_72043781423501
// MI455X (gfx1250) — matrix operand order repaired and checked by reading, not compiled, not run on hardware
//
#include <hip/hip_runtime.h>

// ---------------------------------------------------------------------------
// AlbertDecoderAttention for MI455X (gfx1250, wave32, WMMA)
// bf16 tensor-core GEMMs + flash attention with online softmax.
// ---------------------------------------------------------------------------

#define HID_ 1024
#define H_   16
#define DH_  64
#define B_   4
#define T_   1024
#define S_   1024

typedef __bf16 bf16;
typedef __attribute__((ext_vector_type(16))) __bf16 v16bf;
typedef __attribute__((ext_vector_type(8)))  __bf16 v8bf;
typedef __attribute__((ext_vector_type(8)))  float  v8f;

// ---- WMMA fragment loaders (CDNA5 16x16x32 bf16 layouts) -------------------
// A (16xK=32, bf16): lane 0-15 -> M=lane, VGPR0..3 = K {0..7}+half*8,
//                    VGPR4..7 = K {16..23}+half*8  (half = lane>=16)
// => per lane: two contiguous 8-element (16B) loads from row-major A.
static __device__ __forceinline__ v16bf load_afrag(const bf16* base, int lda,
                                                   int k0, int lane) {
  const int half = lane >> 4;
  const int m    = lane & 15;
  const bf16* p  = base + (size_t)m * lda + k0 + half * 8;
  v8bf lo = *(const v8bf*)p;
  v8bf hi = *(const v8bf*)(p + 16);
  v16bf r;
#pragma unroll
  for (int i = 0; i < 8; ++i) { r[i] = lo[i]; r[i + 8] = hi[i]; }
  return r;
}

// B (K=32 x 16, bf16): lane 0-15 -> N=lane, K {0..7}+half*8 then {16..23}+half*8
// (same K order as A). With the operand stored row-major as [N][K], each lane
// needs two contiguous 8-element (16B) loads.
static __device__ __forceinline__ v16bf load_bfrag(const bf16* base, int ldn,
                                                   int k0, int lane) {
  const int half = lane >> 4;
  const int n    = lane & 15;
  const bf16* p  = base + (size_t)n * ldn + k0 + half * 8;
  v8bf lo = *(const v8bf*)p;
  v8bf hi = *(const v8bf*)(p + 16);
  v16bf r;
#pragma unroll
  for (int i = 0; i < 8; ++i) { r[i] = lo[i]; r[i + 8] = hi[i]; }
  return r;
}

static __device__ __forceinline__ v8f vzero8() {
  v8f z = {0.f, 0.f, 0.f, 0.f, 0.f, 0.f, 0.f, 0.f};
  return z;
}

// ---- elementwise helpers ---------------------------------------------------
__global__ void __launch_bounds__(256)
k_cast_bf16(const float* __restrict__ x, bf16* __restrict__ y, int n) {
  int i = blockIdx.x * 256 + threadIdx.x;
  if (i < n) y[i] = (bf16)x[i];
}

// W[K,N] fp32 -> WT[N,K] bf16 (coalesced read)
__global__ void __launch_bounds__(256)
k_wtrans(const float* __restrict__ W, bf16* __restrict__ WT, int K, int N) {
  int i = blockIdx.x * 256 + threadIdx.x;
  if (i < K * N) {
    int n = i % N;
    int k = i / N;
    WT[(size_t)n * K + k] = (bf16)W[i];
  }
}

// V [B,Tok,HID] bf16 -> VT [B,H,DH,Tok] bf16
__global__ void __launch_bounds__(256)
k_vtrans(const bf16* __restrict__ V, bf16* __restrict__ VT, int Tok) {
  int i   = blockIdx.x * 256 + threadIdx.x;  // exact multiple of grid
  int hid = i % HID_;
  int bt  = i / HID_;
  int tt  = bt % Tok;
  int b   = bt / Tok;
  int h   = hid >> 6;
  int d   = hid & 63;
  VT[(((size_t)b * H_ + h) * DH_ + d) * Tok + tt] = V[i];
}

// ---- GEMM: out = A(bf16)[M,K] x WT(bf16)[N,K]^T + bias (+residual) ---------
// One wave computes a 16(M) x 64(N) strip; A-frag reused across 4 N-tiles.
__global__ void __launch_bounds__(256)
k_gemm_wmma(const bf16* __restrict__ A, const bf16* __restrict__ WT,
            const float* __restrict__ bias, const float* __restrict__ residual,
            float* __restrict__ outF, bf16* __restrict__ outB,
            int M, int N, int K) {
  const int lane    = threadIdx.x & 31;
  const int wid     = (blockIdx.x * blockDim.x + threadIdx.x) >> 5;
  const int nStrips = N >> 6;
  const int im      = wid / nStrips;
  const int in      = wid - im * nStrips;
  if (im >= (M >> 4)) return;
  const int m0 = im << 4;
  const int n0 = in << 6;

  v8f c[4];
#pragma unroll
  for (int g = 0; g < 4; ++g) c[g] = vzero8();

  const bf16* Abase = A + (size_t)m0 * K;
  for (int k = 0; k < K; k += 32) {
    v16bf a = load_afrag(Abase, K, k, lane);
#pragma unroll
    for (int g = 0; g < 4; ++g) {
      v16bf b = load_bfrag(WT + (size_t)(n0 + g * 16) * K, K, k, lane);
      c[g] = __builtin_amdgcn_wmma_f32_16x16x32_bf16(false, a, false, b,
                                                     (short)0, c[g], false, false);
    }
  }

  const int half = lane >> 4;
  const int nl   = lane & 15;
#pragma unroll
  for (int g = 0; g < 4; ++g) {
#pragma unroll
    for (int v = 0; v < 8; ++v) {
      int    row = m0 + half * 8 + v;   // C layout: lanes16-31 hold M=v+8
      int    col = n0 + g * 16 + nl;
      size_t idx = (size_t)row * N + col;
      float  val = c[g][v] + bias[col];
      if (residual) val += residual[idx];
      if (outF) outF[idx] = val;
      if (outB) outB[idx] = (bf16)val;
    }
  }
}

// ---- Flash attention: one wave per (b, h, 16-query tile) -------------------
// Q,K: [B,Tq/Tk,HID] bf16 (head slice 64 wide). VT: [B,H,DH,Tk] bf16.
// ctx: [B,Tq,HID] bf16.  mask: [B,1,1,Tk] fp32.
__global__ void __launch_bounds__(256)
k_flash_wmma(const bf16* __restrict__ Q, const bf16* __restrict__ Kt,
             const bf16* __restrict__ VT, const float* __restrict__ mask,
             bf16* __restrict__ ctx, int Tq, int Tk) {
  __shared__ bf16 plds[8 * 16 * 32];           // per-wave 16x32 P tile
  const int lane  = threadIdx.x & 31;
  const int wslot = threadIdx.x >> 5;
  bf16* pw = plds + wslot * (16 * 32);

  const int wid    = blockIdx.x * 8 + wslot;
  const int qTiles = Tq >> 4;
  const int qt = wid % qTiles;
  const int h  = (wid / qTiles) % H_;
  const int b  = wid / (qTiles * H_);

  const bf16* Qbase = Q + ((size_t)(b * Tq + qt * 16)) * HID_ + h * DH_;
  v16bf qa0 = load_afrag(Qbase, HID_, 0, lane);    // K(dh) = 0..31
  v16bf qa1 = load_afrag(Qbase, HID_, 32, lane);   // K(dh) = 32..63

  const bf16*  Kbase = Kt + (size_t)b * Tk * HID_ + h * DH_;
  const bf16*  Vbase = VT + ((size_t)(b * H_ + h)) * DH_ * Tk;
  const float* mbase = mask + (size_t)b * Tk;

  float mrun[8], lrun[8];
  v8f   acc[4];
#pragma unroll
  for (int v = 0; v < 8; ++v) { mrun[v] = -1e30f; lrun[v] = 0.0f; }
#pragma unroll
  for (int g = 0; g < 4; ++g) acc[g] = vzero8();

  const int   half  = lane >> 4;
  const int   nl    = lane & 15;
  const float scale = 0.125f;  // 1/sqrt(DH=64)

  for (int j0 = 0; j0 < Tk; j0 += 32) {
    // ---- S = Q . K^T over 32 keys -> two 16x16 fp32 tiles
    v8f cL = vzero8(), cR = vzero8();
    {
      const bf16* KL = Kbase + (size_t)j0 * HID_;        // keys j0..j0+15
      const bf16* KR = Kbase + (size_t)(j0 + 16) * HID_; // keys j0+16..j0+31
      v16bf bL0 = load_bfrag(KL, HID_, 0, lane);
      v16bf bL1 = load_bfrag(KL, HID_, 32, lane);
      v16bf bR0 = load_bfrag(KR, HID_, 0, lane);
      v16bf bR1 = load_bfrag(KR, HID_, 32, lane);
      cL = __builtin_amdgcn_wmma_f32_16x16x32_bf16(false, qa0, false, bL0, (short)0, cL, false, false);
      cL = __builtin_amdgcn_wmma_f32_16x16x32_bf16(false, qa1, false, bL1, (short)0, cL, false, false);
      cR = __builtin_amdgcn_wmma_f32_16x16x32_bf16(false, qa0, false, bR0, (short)0, cR, false, false);
      cR = __builtin_amdgcn_wmma_f32_16x16x32_bf16(false, qa1, false, bR1, (short)0, cR, false, false);
    }
    const float mL = mbase[j0 + nl];
    const float mR = mbase[j0 + 16 + nl];

    // ---- online softmax: rows live as (VGPR v, half) = row half*8+v; the
    //      16 key columns of a row live across one 16-lane half.
#pragma unroll
    for (int v = 0; v < 8; ++v) {
      float sL = cL[v] * scale + mL;
      float sR = cR[v] * scale + mR;
      float t  = fmaxf(sL, sR);
#pragma unroll
      for (int o = 8; o >= 1; o >>= 1) t = fmaxf(t, __shfl_xor(t, o, 16));
      float mn    = fmaxf(mrun[v], t);
      float alpha = __expf(mrun[v] - mn);
      mrun[v] = mn;
      float pl = __expf(sL - mn);
      float pr = __expf(sR - mn);
      float rs = pl + pr;
#pragma unroll
      for (int o = 8; o >= 1; o >>= 1) rs += __shfl_xor(rs, o, 16);
      lrun[v] = lrun[v] * alpha + rs;
#pragma unroll
      for (int g = 0; g < 4; ++g) acc[g][v] *= alpha;
      // stash P (bf16) into per-wave LDS, row-major [16][32]
      const int row = half * 8 + v;
      pw[row * 32 + nl]      = (bf16)pl;
      pw[row * 32 + 16 + nl] = (bf16)pr;
    }
    // DS ops of a wave are in-order; wait for LDS stores before A-frag reads.
    asm volatile("s_wait_dscnt 0" ::: "memory");
    v16bf pa = load_afrag(pw, 32, 0, lane);  // P as 16x32 A-operand

    // ---- acc += P . V  (V^T gives contiguous B-frags along keys)
#pragma unroll
    for (int g = 0; g < 4; ++g) {
      v16bf vb = load_bfrag(Vbase + (size_t)(g * 16) * Tk + j0, Tk, 0, lane);
      acc[g] = __builtin_amdgcn_wmma_f32_16x16x32_bf16(false, pa, false, vb,
                                                       (short)0, acc[g], false, false);
    }
  }

  // ---- normalize and store ctx tile
#pragma unroll
  for (int v = 0; v < 8; ++v) {
    float  inv   = 1.0f / lrun[v];
    int    row   = half * 8 + v;
    size_t rbase = ((size_t)(b * Tq + qt * 16 + row)) * HID_ + h * DH_;
#pragma unroll
    for (int g = 0; g < 4; ++g)
      ctx[rbase + g * 16 + nl] = (bf16)(acc[g][v] * inv);
  }
}

// ---- LayerNorm over HID=1024, one block per row ----------------------------
__global__ void __launch_bounds__(256)
k_layernorm(const float* __restrict__ x, const float* __restrict__ gamma,
            const float* __restrict__ beta, float* __restrict__ outF,
            bf16* __restrict__ outB) {
  __shared__ float red[256];
  const int row = blockIdx.x;
  const int t   = threadIdx.x;
  const float* xr = x + (size_t)row * HID_;
  float v[4];
#pragma unroll
  for (int j = 0; j < 4; ++j) v[j] = xr[t + j * 256];

  red[t] = v[0] + v[1] + v[2] + v[3];
  __syncthreads();
  for (int o = 128; o > 0; o >>= 1) {
    if (t < o) red[t] += red[t + o];
    __syncthreads();
  }
  const float mu = red[0] * (1.0f / HID_);
  __syncthreads();

  float d0 = v[0] - mu, d1 = v[1] - mu, d2 = v[2] - mu, d3 = v[3] - mu;
  red[t] = d0 * d0 + d1 * d1 + d2 * d2 + d3 * d3;
  __syncthreads();
  for (int o = 128; o > 0; o >>= 1) {
    if (t < o) red[t] += red[t + o];
    __syncthreads();
  }
  const float var = red[0] * (1.0f / HID_);
  const float inv = rsqrtf(var + 1e-12f);
#pragma unroll
  for (int j = 0; j < 4; ++j) {
    int   cidx = t + j * 256;
    float y    = (v[j] - mu) * inv * gamma[cidx] + beta[cidx];
    if (outF) outF[(size_t)row * HID_ + cidx] = y;
    if (outB) outB[(size_t)row * HID_ + cidx] = (bf16)y;
  }
}

// ---------------------------------------------------------------------------
extern "C" void kernel_launch(void* const* d_in, const int* in_sizes, int n_in,
                              void* d_out, int out_size, void* d_ws, size_t ws_size,
                              hipStream_t stream) {
  (void)in_sizes; (void)n_in; (void)out_size; (void)ws_size;
  const float* enc  = (const float*)d_in[0];
  const float* dec  = (const float*)d_in[1];
  const float* srcm = (const float*)d_in[2];
  const float* tgtm = (const float*)d_in[3];
  const float* Wq   = (const float*)d_in[4];  const float* bq  = (const float*)d_in[5];
  const float* Wk   = (const float*)d_in[6];  const float* bk  = (const float*)d_in[7];
  const float* Wv   = (const float*)d_in[8];  const float* bv  = (const float*)d_in[9];
  const float* Wq2  = (const float*)d_in[10]; const float* bq2 = (const float*)d_in[11];
  const float* Wk2  = (const float*)d_in[12]; const float* bk2 = (const float*)d_in[13];
  const float* Wv2  = (const float*)d_in[14]; const float* bv2 = (const float*)d_in[15];
  const float* Wo   = (const float*)d_in[16]; const float* bo  = (const float*)d_in[17];
  const float* gam  = (const float*)d_in[18]; const float* bet = (const float*)d_in[19];

  const int    M    = B_ * T_;                                // 4096 rows
  const size_t actB = (size_t)M * HID_ * sizeof(bf16);        // 8 MB
  const size_t actF = (size_t)M * HID_ * sizeof(float);       // 16 MB
  const size_t wB   = (size_t)HID_ * HID_ * sizeof(bf16);     // 2 MB

  char*  p   = (char*)d_ws;
  size_t off = 0;
  auto take = [&](size_t bytes) -> char* {
    char* r = p + off;
    off = (off + bytes + 255) & ~(size_t)255;
    return r;
  };
  bf16*  xb   = (bf16*)take(actB);   // casted activations (dec, then enc)
  bf16*  qb   = (bf16*)take(actB);   // q projection (bf16)
  bf16*  kb   = (bf16*)take(actB);
  bf16*  vb   = (bf16*)take(actB);
  bf16*  vt   = (bf16*)take(actB);   // V^T [B,H,DH,Tk]
  bf16*  ctxb = (bf16*)take(actB);
  bf16*  sob  = (bf16*)take(actB);   // self_out (bf16, feeds q2 projection)
  float* qf   = (float*)take(actF);  // q projection fp32 (residual)
  float* pref = (float*)take(actF);  // pre-LayerNorm fp32
  bf16*  WqT  = (bf16*)take(wB);
  bf16*  WkT  = (bf16*)take(wB);
  bf16*  WvT  = (bf16*)take(wB);
  bf16*  Wq2T = (bf16*)take(wB);
  bf16*  Wk2T = (bf16*)take(wB);
  bf16*  Wv2T = (bf16*)take(wB);
  bf16*  WoT  = (bf16*)take(wB);

  const int nAct   = M * HID_;                         // 4,194,304
  const int gAct   = nAct / 256;                       // 16384
  const int gW     = (HID_ * HID_) / 256;              // 4096
  const int gGemm  = ((M / 16) * (HID_ / 64)) / 8;     // 512 blocks (8 waves)
  const int gFlash = (B_ * H_ * (T_ / 16)) / 8;        // 512 blocks

  // weights -> bf16 transposed [N,K]
  k_wtrans<<<gW, 256, 0, stream>>>(Wq,  WqT,  HID_, HID_);
  k_wtrans<<<gW, 256, 0, stream>>>(Wk,  WkT,  HID_, HID_);
  k_wtrans<<<gW, 256, 0, stream>>>(Wv,  WvT,  HID_, HID_);
  k_wtrans<<<gW, 256, 0, stream>>>(Wq2, Wq2T, HID_, HID_);
  k_wtrans<<<gW, 256, 0, stream>>>(Wk2, Wk2T, HID_, HID_);
  k_wtrans<<<gW, 256, 0, stream>>>(Wv2, Wv2T, HID_, HID_);
  k_wtrans<<<gW, 256, 0, stream>>>(Wo,  WoT,  HID_, HID_);  // [H*DH, HID] flat

  // ---------------- self-attention ----------------
  k_cast_bf16<<<gAct, 256, 0, stream>>>(dec, xb, nAct);
  k_gemm_wmma<<<gGemm, 256, 0, stream>>>(xb, WqT, bq, nullptr, qf, qb, M, HID_, HID_);
  k_gemm_wmma<<<gGemm, 256, 0, stream>>>(xb, WkT, bk, nullptr, nullptr, kb, M, HID_, HID_);
  k_gemm_wmma<<<gGemm, 256, 0, stream>>>(xb, WvT, bv, nullptr, nullptr, vb, M, HID_, HID_);
  k_vtrans<<<gAct, 256, 0, stream>>>(vb, vt, T_);
  k_flash_wmma<<<gFlash, 256, 0, stream>>>(qb, kb, vt, tgtm, ctxb, T_, T_);
  // out-proj + bias + residual(q1) -> pre-LN
  k_gemm_wmma<<<gGemm, 256, 0, stream>>>(ctxb, WoT, bo, qf, pref, nullptr, M, HID_, HID_);
  k_layernorm<<<M, 256, 0, stream>>>(pref, gam, bet, nullptr, sob);

  // ---------------- cross-attention ----------------
  k_gemm_wmma<<<gGemm, 256, 0, stream>>>(sob, Wq2T, bq2, nullptr, qf, qb, M, HID_, HID_);
  k_cast_bf16<<<gAct, 256, 0, stream>>>(enc, xb, nAct);
  k_gemm_wmma<<<gGemm, 256, 0, stream>>>(xb, Wk2T, bk2, nullptr, nullptr, kb, M, HID_, HID_);
  k_gemm_wmma<<<gGemm, 256, 0, stream>>>(xb, Wv2T, bv2, nullptr, nullptr, vb, M, HID_, HID_);
  k_vtrans<<<gAct, 256, 0, stream>>>(vb, vt, S_);
  k_flash_wmma<<<gFlash, 256, 0, stream>>>(qb, kb, vt, srcm, ctxb, T_, S_);
  k_gemm_wmma<<<gGemm, 256, 0, stream>>>(ctxb, WoT, bo, qf, pref, nullptr, M, HID_, HID_);
  k_layernorm<<<M, 256, 0, stream>>>(pref, gam, bet, (float*)d_out, nullptr);
}
